// GptOssAttention_42752104464388
// MI455X (gfx1250) — matrix operand order repaired and checked by reading, not compiled, not run on hardware
//
#include <hip/hip_runtime.h>
#include <hip/hip_bf16.h>
#include <stdint.h>

typedef __attribute__((ext_vector_type(16))) __bf16 v16bf;
typedef __attribute__((ext_vector_type(8)))  __bf16 v8bf;
typedef __attribute__((ext_vector_type(8)))  float  v8f;

#define S_LEN 2048
#define HIDN  2880
#define NH    64
#define NKV   8
#define DH    64
#define WIN   128
#define NQKV  5120            // 4096 q + 512 k + 512 v
#define KOFF  4096
#define VOFF  4608
#define SPAD  2080
#define QK_SCALE 0.125f
#define NEGINF (-1e30f)

__device__ __forceinline__ __bf16 f2bf(float f) {
  uint32_t u; __builtin_memcpy(&u, &f, 4);
  uint32_t r = u + 0x7FFFu + ((u >> 16) & 1u);
  uint16_t h = (uint16_t)(r >> 16);
  __bf16 b; __builtin_memcpy(&b, &h, 2);
  return b;
}
__device__ __forceinline__ float bf2f(__bf16 v) {
  uint16_t h; __builtin_memcpy(&h, &v, 2);
  uint32_t u = ((uint32_t)h) << 16;
  float f; __builtin_memcpy(&f, &u, 4);
  return f;
}
__device__ __forceinline__ v8f wmma_bf16(v16bf a, v16bf b, v8f c) {
  return __builtin_amdgcn_wmma_f32_16x16x32_bf16(false, a, false, b, (short)0, c, false, false);
}
__device__ __forceinline__ v16bf cat8(v8bf lo, v8bf hi) {
  return __builtin_shufflevector(lo, hi, 0,1,2,3,4,5,6,7,8,9,10,11,12,13,14,15);
}

// ---------------- pack kernels ----------------
__global__ void k_pack_x(const float* __restrict__ x, __bf16* __restrict__ xb, int n) {
  int i = blockIdx.x * blockDim.x + threadIdx.x;
  if (i < n) xb[i] = f2bf(x[i]);
}

__global__ void k_pack_wqkv(const float* __restrict__ Wq, const float* __restrict__ Wk,
                            const float* __restrict__ Wv, const float* __restrict__ bq,
                            const float* __restrict__ bk, const float* __restrict__ bv,
                            __bf16* __restrict__ Wt, float* __restrict__ bcat, int n) {
  int i = blockIdx.x * blockDim.x + threadIdx.x;
  if (i >= n) return;
  int k = i % HIDN;
  int c = i / HIDN;                      // output column 0..5119
  float w;
  if (c < KOFF)      w = Wq[(size_t)k * 4096 + c];
  else if (c < VOFF) w = Wk[(size_t)k * 512 + (c - KOFF)];
  else               w = Wv[(size_t)k * 512 + (c - VOFF)];
  Wt[i] = f2bf(w);
  if (k == 0) {
    float b = (c < KOFF) ? bq[c] : ((c < VOFF) ? bk[c - KOFF] : bv[c - VOFF]);
    bcat[c] = b;
  }
}

__global__ void k_pack_wo(const float* __restrict__ Wo, __bf16* __restrict__ WoT, int n) {
  int i = blockIdx.x * blockDim.x + threadIdx.x;
  if (i >= n) return;
  int k = i % 4096;
  int c = i / 4096;                       // 0..2879
  WoT[i] = f2bf(Wo[(size_t)k * HIDN + c]);
}

// ---------------- GEMM core: one wave -> 32x64 tile (2 M-subtiles share B) ----------------
__device__ __forceinline__ void gemm_tile2(const __bf16* __restrict__ A, int lda,
                                           const __bf16* __restrict__ Bt, int ldb,
                                           int K, int m0, int n0, int lane, v8f acc[8]) {
  const int hf = lane >> 4;
  const int lr = lane & 15;
  const __bf16* arow0 = A + (size_t)(m0 + lr) * lda + 16 * hf;
  const __bf16* arow1 = arow0 + (size_t)16 * lda;
  const __bf16* br[4];
#pragma unroll
  for (int t = 0; t < 4; ++t)
    br[t] = Bt + (size_t)(n0 + t * 16 + lr) * ldb + 16 * hf;
  for (int k0 = 0; k0 < K; k0 += 32) {
    v16bf a0 = cat8(*(const v8bf*)(arow0 + k0), *(const v8bf*)(arow0 + k0 + 8));
    v16bf a1 = cat8(*(const v8bf*)(arow1 + k0), *(const v8bf*)(arow1 + k0 + 8));
#pragma unroll
    for (int t = 0; t < 4; ++t) {
      v16bf bv = *(const v16bf*)(br[t] + k0);
      acc[t]     = wmma_bf16(a0, bv, acc[t]);
      acc[4 + t] = wmma_bf16(a1, bv, acc[4 + t]);
    }
  }
}

__global__ __launch_bounds__(32) void k_gemm_qkv(const __bf16* __restrict__ A,
                                                 const __bf16* __restrict__ Bt,
                                                 const float* __restrict__ bias,
                                                 __bf16* __restrict__ C) {
  const int m0 = blockIdx.x * 32, n0 = blockIdx.y * 64;
  const int lane = threadIdx.x, hf = lane >> 4, lr = lane & 15;
  v8f acc[8] = {};
  gemm_tile2(A, HIDN, Bt, HIDN, HIDN, m0, n0, lane, acc);
#pragma unroll
  for (int t = 0; t < 4; ++t) {
    int col = n0 + t * 16 + lr;
    float bs = bias[col];
#pragma unroll
    for (int r = 0; r < 8; ++r) {
      int row = m0 + r + 8 * hf;
      C[(size_t)row * NQKV + col]        = f2bf(acc[t][r] + bs);
      C[(size_t)(row + 16) * NQKV + col] = f2bf(acc[4 + t][r] + bs);
    }
  }
}

__global__ __launch_bounds__(32) void k_gemm_o(const __bf16* __restrict__ A,
                                               const __bf16* __restrict__ Bt,
                                               const float* __restrict__ bias,
                                               float* __restrict__ C) {
  const int m0 = blockIdx.x * 32, n0 = blockIdx.y * 64;
  const int lane = threadIdx.x, hf = lane >> 4, lr = lane & 15;
  v8f acc[8] = {};
  gemm_tile2(A, NH * DH, Bt, NH * DH, NH * DH, m0, n0, lane, acc);
#pragma unroll
  for (int t = 0; t < 4; ++t) {
    int col = n0 + t * 16 + lr;
    float bs = bias[col];
#pragma unroll
    for (int r = 0; r < 8; ++r) {
      int row = m0 + r + 8 * hf;
      C[(size_t)row * HIDN + col]        = acc[t][r] + bs;
      C[(size_t)(row + 16) * HIDN + col] = acc[4 + t][r] + bs;
    }
  }
}

// ---------------- RoPE (in place on q,k columns of qkv) ----------------
__global__ void k_rope(__bf16* __restrict__ qkv, const float* __restrict__ cosb,
                       const float* __restrict__ sinb, int n) {
  int i = blockIdx.x * blockDim.x + threadIdx.x;
  if (i >= n) return;                     // n = S * 72 * 32
  int d2 = i & 31;
  int t  = i >> 5;
  int head = t % 72;                      // 0..63 q heads, 64..71 k heads
  int s = t / 72;
  __bf16* p = qkv + (size_t)s * NQKV + head * 64 + 2 * d2;
  float c  = cosb[s * 32 + d2];
  float sn = sinb[s * 32 + d2];
  float x1 = bf2f(p[0]);
  float x2 = bf2f(p[1]);
  p[0] = f2bf(x1 * c - x2 * sn);
  p[1] = f2bf(x1 * sn + x2 * c);
}

// ---------------- V transpose: vT[hk][d][s] (padded, zeroed tail) ----------------
__global__ void k_vtrans(const __bf16* __restrict__ qkv, __bf16* __restrict__ vT, int n) {
  int i = blockIdx.x * blockDim.x + threadIdx.x;
  if (i >= n) return;                     // n = 8 * 64 * SPAD
  int s = i % SPAD;
  int t = i / SPAD;
  int d = t % 64;
  int hk = t / 64;
  __bf16 val = f2bf(0.0f);
  if (s < S_LEN) val = qkv[(size_t)s * NQKV + VOFF + hk * 64 + d];
  vT[i] = val;
}

// ---------------- attention: 1 wave = (16 queries x 1 head) ----------------
__global__ __launch_bounds__(32) void k_attn(const __bf16* __restrict__ qkv,
                                             const __bf16* __restrict__ vT,
                                             const float* __restrict__ sinks,
                                             __bf16* __restrict__ out) {
  __shared__ __align__(32) __bf16 plds[16 * 32];
  const int i0 = blockIdx.x * 16;
  const int h  = blockIdx.y;
  const int hk = h >> 3;
  const int lane = threadIdx.x, hf = lane >> 4, lr = lane & 15;

  // zero the upper-16-keys half of the P tile (A operand padding)
  for (int idx = lane; idx < 256; idx += 32)
    plds[(idx >> 4) * 32 + 16 + (idx & 15)] = f2bf(0.0f);

  // Q in A layout (rows i0+lr, K chunk 16*hf..16*hf+15 within each 32-step)
  const __bf16* qbase = qkv + (size_t)(i0 + lr) * NQKV + h * DH;
  v16bf qa[2];
#pragma unroll
  for (int c = 0; c < 2; ++c) {
    const __bf16* p = qbase + 32 * c + 16 * hf;
    qa[c] = cat8(*(const v8bf*)p, *(const v8bf*)(p + 8));
  }

  const float snk = sinks[h];
  float mrow[8], lrow[8];
#pragma unroll
  for (int r = 0; r < 8; ++r) { mrow[r] = snk; lrow[r] = 1.0f; }
  v8f o[4] = {};

  const int jb0 = (i0 >= WIN) ? ((i0 - WIN + 1) >> 4) : 0;
  const int jb1 = i0 >> 4;
  for (int jb = jb0; jb <= jb1; ++jb) {
    const int j0 = jb * 16;
    // scores: Q(16x64) x K^T(64x16)
    v8f sc = {};
#pragma unroll
    for (int c = 0; c < 2; ++c) {
      const __bf16* bp = qkv + (size_t)(j0 + lr) * NQKV + KOFF + hk * DH + 32 * c + 16 * hf;
      v16bf bv = *(const v16bf*)bp;
      sc = wmma_bf16(qa[c], bv, sc);
    }
    const int j = j0 + lr;                // this lane's key column
    float pv[8], rscale[8];
#pragma unroll
    for (int r = 0; r < 8; ++r) {
      int qi = i0 + r + 8 * hf;
      float z = ((j <= qi) && (qi - j < WIN)) ? sc[r] * QK_SCALE : NEGINF;
      float red = z;
      red = fmaxf(red, __shfl_xor(red, 1, 32));
      red = fmaxf(red, __shfl_xor(red, 2, 32));
      red = fmaxf(red, __shfl_xor(red, 4, 32));
      red = fmaxf(red, __shfl_xor(red, 8, 32));
      float mnew = fmaxf(mrow[r], red);
      float scl  = __expf(mrow[r] - mnew);
      mrow[r] = mnew;
      float p = __expf(z - mnew);
      float ps = p;
      ps += __shfl_xor(ps, 1, 32);
      ps += __shfl_xor(ps, 2, 32);
      ps += __shfl_xor(ps, 4, 32);
      ps += __shfl_xor(ps, 8, 32);
      lrow[r] = lrow[r] * scl + ps;
      rscale[r] = scl;
      pv[r] = p;
    }
#pragma unroll
    for (int t = 0; t < 4; ++t)
#pragma unroll
      for (int r = 0; r < 8; ++r) o[t][r] *= rscale[r];

    // C-layout -> A-layout via LDS
#pragma unroll
    for (int r = 0; r < 8; ++r)
      plds[(r + 8 * hf) * 32 + lr] = f2bf(pv[r]);
    asm volatile("s_wait_dscnt 0" ::: "memory");
    const __bf16* pl = plds + lr * 32 + 16 * hf;
    v16bf pa = cat8(*(const v8bf*)pl, *(const v8bf*)(pl + 8));

    // P(16x32, upper half zero) x V(32x64)
#pragma unroll
    for (int t = 0; t < 4; ++t) {
      const __bf16* vb = vT + (size_t)(hk * DH + t * 16 + lr) * SPAD + j0 + 16 * hf;
      v16bf bv = *(const v16bf*)vb;
      o[t] = wmma_bf16(pa, bv, o[t]);
    }
    asm volatile("s_wait_dscnt 0" ::: "memory");
  }

#pragma unroll
  for (int t = 0; t < 4; ++t) {
    int col = h * DH + t * 16 + lr;
#pragma unroll
    for (int r = 0; r < 8; ++r) {
      int row = i0 + r + 8 * hf;
      out[(size_t)row * (NH * DH) + col] = f2bf(o[t][r] / lrow[r]);
    }
  }
}

// ---------------- host ----------------
extern "C" void kernel_launch(void* const* d_in, const int* in_sizes, int n_in,
                              void* d_out, int out_size, void* d_ws, size_t ws_size,
                              hipStream_t stream) {
  (void)in_sizes; (void)n_in; (void)out_size; (void)ws_size;
  const float* x    = (const float*)d_in[0];
  const float* cosb = (const float*)d_in[1];
  const float* sinb = (const float*)d_in[2];
  const float* Wq   = (const float*)d_in[3];
  const float* bq   = (const float*)d_in[4];
  const float* Wk   = (const float*)d_in[5];
  const float* bk   = (const float*)d_in[6];
  const float* Wv   = (const float*)d_in[7];
  const float* bv   = (const float*)d_in[8];
  const float* Wo   = (const float*)d_in[9];
  const float* bo   = (const float*)d_in[10];
  const float* sinks = (const float*)d_in[11];
  float* out = (float*)d_out;

  char* ws = (char*)d_ws;
  size_t off = 0;
  auto take = [&](size_t bytes) -> void* {
    void* p = ws + off;
    off = (off + bytes + 255) & ~((size_t)255);
    return p;
  };
  __bf16* xb    = (__bf16*)take((size_t)S_LEN * HIDN * 2);
  __bf16* WcatT = (__bf16*)take((size_t)NQKV * HIDN * 2);
  float*  bcat  = (float*) take((size_t)NQKV * 4);
  __bf16* qkv   = (__bf16*)take((size_t)S_LEN * NQKV * 2);
  __bf16* WoT   = (__bf16*)take((size_t)HIDN * 4096 * 2);
  __bf16* vT    = (__bf16*)take((size_t)NKV * DH * SPAD * 2);
  __bf16* aout  = (__bf16*)take((size_t)S_LEN * NH * DH * 2);

  const int T = 256;
  {
    int n = S_LEN * HIDN;
    k_pack_x<<<(n + T - 1) / T, T, 0, stream>>>(x, xb, n);
  }
  {
    int n = NQKV * HIDN;
    k_pack_wqkv<<<(n + T - 1) / T, T, 0, stream>>>(Wq, Wk, Wv, bq, bk, bv, WcatT, bcat, n);
  }
  {
    int n = HIDN * 4096;
    k_pack_wo<<<(n + T - 1) / T, T, 0, stream>>>(Wo, WoT, n);
  }
  k_gemm_qkv<<<dim3(S_LEN / 32, NQKV / 64), 32, 0, stream>>>(xb, WcatT, bcat, qkv);
  {
    int n = S_LEN * (NH + NKV) * (DH / 2);
    k_rope<<<(n + T - 1) / T, T, 0, stream>>>(qkv, cosb, sinb, n);
  }
  {
    int n = NKV * DH * SPAD;
    k_vtrans<<<(n + T - 1) / T, T, 0, stream>>>(qkv, vT, n);
  }
  k_attn<<<dim3(S_LEN / 16, NH), 32, 0, stream>>>(qkv, vT, sinks, aout);
  k_gemm_o<<<dim3(S_LEN / 32, HIDN / 64), 32, 0, stream>>>(aout, WoT, bo, out);
}
